// HGTModel_51333449122063
// MI455X (gfx1250) — compile-verified, matrix operand order checked by reading, not run on hardware
//
#include <hip/hip_runtime.h>
#include <hip/hip_fp16.h>

#define NN 50000
#define EE 400000
#define D  128
#define NTY 3
#define NET 4
#define NH 8
#define HD 16

#define KEY_NEG_INF 0x007FFFFFu   // fkey(-inf) = ~0xFF800000

typedef __attribute__((ext_vector_type(16))) _Float16 v16h;
typedef __attribute__((ext_vector_type(8)))  float    v8f;

union AFrag { v16h v; _Float16 h[16]; unsigned u[8]; };
union CFrag { v8f v; float f[8]; };

__device__ __forceinline__ unsigned fkey(float f) {
  unsigned u = __float_as_uint(f);
  return (u & 0x80000000u) ? ~u : (u | 0x80000000u);
}
__device__ __forceinline__ float fdec(unsigned k) {
  unsigned u = (k & 0x80000000u) ? (k & 0x7fffffffu) : ~k;
  return __uint_as_float(u);
}

__global__ void k_f32_to_f16(const float* __restrict__ in, _Float16* __restrict__ out, int n) {
  int i = blockIdx.x * 256 + threadIdx.x;
  if (i < n) out[i] = (_Float16)in[i];
}

__global__ void k_fill_u32(unsigned* __restrict__ p, unsigned val, int n) {
  int i = blockIdx.x * 256 + threadIdx.x;
  if (i < n) p[i] = val;
}

// Pack k-major f32 weights [nmat][128][128] into WMMA B-fragment lane layout:
// Wp[((m*8 + col_tile)*4 + k_chunk)*512 + lane*16 + j]  (f16)
// where j=2r+h, K = ((r<4)?2r:16+2(r-4)) + h + 8*(lane>>4) + k_chunk*32, col = col_tile*16 + (lane&15)
__global__ void k_pack_w(const float* __restrict__ W, _Float16* __restrict__ Wp, int nmat) {
  const int idx = blockIdx.x * 256 + threadIdx.x;
  const int total = nmat * 8 * 4 * 32 * 16;
  if (idx >= total) return;
  const int j    = idx & 15;
  const int lane = (idx >> 4) & 31;
  const int kc   = (idx >> 9) & 3;
  const int ct   = (idx >> 11) & 7;
  const int m    = idx >> 14;
  const int r = j >> 1, h = j & 1, g = lane >> 4, mr = lane & 15;
  const int k = ((r < 4) ? 2 * r : 16 + 2 * (r - 4)) + h + 8 * g + kc * 32;
  const int col = ct * 16 + mr;
  Wp[idx] = (_Float16)W[(size_t)m * D * D + (size_t)k * D + col];
}

// Per-node-type GEMM: Y[n,c] = act( sum_k X[n,k] * W[t(n)][k][c] + b[t(n)][c] )
// W is pre-packed (k_pack_w). MODE 0: tanh, f32+f16. MODE 1: plain f32. MODE 2: HGT residual.
template <int MODE>
__global__ __launch_bounds__(256)
void k_gemm_ts(const _Float16* __restrict__ X,      // [NN, D] f16
               const _Float16* __restrict__ Wp,     // [3][8][4][32][16] f16 packed
               const float* __restrict__ bias,      // [3, D]
               const int* __restrict__ ntype,       // [NN]
               float* __restrict__ out_f32,
               _Float16* __restrict__ out_f16,
               const float* __restrict__ x_res,     // MODE 2: residual input [NN,D]
               const float* __restrict__ skip)      // MODE 2: [3]
{
  const int lane = threadIdx.x & 31;
  const int wave = threadIdx.x >> 5;
  const int tile = blockIdx.x * 8 + wave;          // 3125*8 = 25000 tiles exact
  const int node_base = (tile >> 3) * 16;
  const int ctile     = tile & 7;
  const int col_base  = ctile * 16;
  const int mr = lane & 15;
  const int g  = lane >> 4;

  CFrag acc0, acc1, acc2;
#pragma unroll
  for (int i = 0; i < 8; ++i) { acc0.f[i] = 0.f; acc1.f[i] = 0.f; acc2.f[i] = 0.f; }

  const _Float16* Xrow = X + (size_t)(node_base + mr) * D;

#pragma unroll
  for (int kc = 0; kc < 4; ++kc) {
    const int kb = kc * 32;
    AFrag a;
#pragma unroll
    for (int r = 0; r < 8; ++r) {
      const int k0 = ((r < 4) ? (2 * r) : (16 + 2 * (r - 4))) + 8 * g + kb;
      a.u[r] = *(const unsigned*)(Xrow + k0);      // two contiguous halfs (K, K+1)
    }
#pragma unroll
    for (int t = 0; t < NTY; ++t) {
      // one contiguous 32B per-lane load; 512 consecutive halfs per wave-fragment
      AFrag b;
      b.v = *(const v16h*)(Wp + ((((size_t)t * 8 + ctile) * 4 + kc) * 32 + lane) * 16);
      if (t == 0)
        acc0.v = __builtin_amdgcn_wmma_f32_16x16x32_f16(false, a.v, false, b.v, (short)0, acc0.v, false, false);
      else if (t == 1)
        acc1.v = __builtin_amdgcn_wmma_f32_16x16x32_f16(false, a.v, false, b.v, (short)0, acc1.v, false, false);
      else
        acc2.v = __builtin_amdgcn_wmma_f32_16x16x32_f16(false, a.v, false, b.v, (short)0, acc2.v, false, false);
    }
  }

  const int col = col_base + mr;
#pragma unroll
  for (int r = 0; r < 8; ++r) {
    const int row = node_base + r + 8 * g;
    const int t = ntype[row];
    float v = (t == 0 ? acc0.f[r] : (t == 1 ? acc1.f[r] : acc2.f[r])) + bias[t * D + col];
    const size_t oi = (size_t)row * D + col;
    if (MODE == 0) {
      v = tanhf(v);
      out_f32[oi] = v;
      out_f16[oi] = (_Float16)v;
    } else if (MODE == 1) {
      out_f32[oi] = v;
    } else {
      const float sc = 1.f / (1.f + expf(-skip[t]));
      const float xv = x_res[oi];
      const float res = v * sc + xv * (1.f - v);
      out_f32[oi] = res;
      out_f16[oi] = (_Float16)res;
    }
  }
}

// Pass 1: att[e,h] = (q[tgt] . (rel_att[r,h]^T k[src])) * pri / 4 ; segment-max via encoded atomicMax
__global__ __launch_bounds__(256)
void k_edge_att(const float* __restrict__ kbuf, const float* __restrict__ qbuf,
                const int* __restrict__ src, const int* __restrict__ tgt,
                const int* __restrict__ etype,
                const float* __restrict__ rel_att,   // [4,8,16,16]
                const float* __restrict__ rel_pri,   // [4,8]
                float* __restrict__ attb, unsigned* __restrict__ nmax)
{
  __shared__ float sA[NET * NH * HD * HD];
  for (int i = threadIdx.x; i < NET * NH * HD * HD; i += 256) sA[i] = rel_att[i];
  __syncthreads();

  const int idx = blockIdx.x * 256 + threadIdx.x;   // 12500 * 256 = EE*NH exact
  const int e = idx >> 3, h = idx & 7;
  const int r = etype[e], s = src[e], t = tgt[e];

  float kk[16], qq[16];
  const float4* kp = (const float4*)(kbuf + (size_t)s * D + h * HD);
  const float4* qp = (const float4*)(qbuf + (size_t)t * D + h * HD);
#pragma unroll
  for (int i = 0; i < 4; ++i) {
    float4 a = kp[i]; kk[4*i] = a.x; kk[4*i+1] = a.y; kk[4*i+2] = a.z; kk[4*i+3] = a.w;
    float4 b = qp[i]; qq[4*i] = b.x; qq[4*i+1] = b.y; qq[4*i+2] = b.z; qq[4*i+3] = b.w;
  }
  const float* A = sA + (r * NH + h) * HD * HD;
  float acc = 0.f;
#pragma unroll
  for (int f = 0; f < HD; ++f) {
    float tmp = 0.f;
#pragma unroll
    for (int d = 0; d < HD; ++d) tmp += kk[d] * A[d * HD + f];
    acc += tmp * qq[f];
  }
  const float att = acc * rel_pri[r * NH + h] * 0.25f;   // 1/sqrt(16)
  attb[idx] = att;
  atomicMax(&nmax[t * NH + h], fkey(att));
}

// Pass 2: ex = exp(att - max); denominator via atomicAdd
__global__ __launch_bounds__(256)
void k_edge_ex(float* __restrict__ attb, const int* __restrict__ tgt,
               const unsigned* __restrict__ nmax, float* __restrict__ den)
{
  const int idx = blockIdx.x * 256 + threadIdx.x;
  const int e = idx >> 3, h = idx & 7;
  const int t = tgt[e];
  float m = fdec(nmax[t * NH + h]);
  if (!isfinite(m)) m = 0.f;
  const float ex = expf(attb[idx] - m);
  attb[idx] = ex;
  atomicAdd(&den[t * NH + h], ex);
}

// Pass 3: alpha = ex/den ; msg = rel_msg[r,h]^T v[src] ; scatter alpha*msg into agg[tgt]
__global__ __launch_bounds__(256)
void k_edge_scatter(const float* __restrict__ vbuf, const float* __restrict__ attb,
                    const int* __restrict__ src, const int* __restrict__ tgt,
                    const int* __restrict__ etype,
                    const float* __restrict__ rel_msg,   // [4,8,16,16]
                    const float* __restrict__ den, float* __restrict__ agg)
{
  __shared__ float sM[NET * NH * HD * HD];
  for (int i = threadIdx.x; i < NET * NH * HD * HD; i += 256) sM[i] = rel_msg[i];
  __syncthreads();

  const int idx = blockIdx.x * 256 + threadIdx.x;
  const int e = idx >> 3, h = idx & 7;
  const int r = etype[e], s = src[e], t = tgt[e];
  const float alpha = attb[idx] / den[t * NH + h];

  float vv[16];
  const float4* vp = (const float4*)(vbuf + (size_t)s * D + h * HD);
#pragma unroll
  for (int i = 0; i < 4; ++i) {
    float4 a = vp[i]; vv[4*i] = a.x; vv[4*i+1] = a.y; vv[4*i+2] = a.z; vv[4*i+3] = a.w;
  }
  const float* M = sM + (r * NH + h) * HD * HD;
  float* aggp = agg + (size_t)t * D + h * HD;
#pragma unroll
  for (int f = 0; f < HD; ++f) {
    float m = 0.f;
#pragma unroll
    for (int d = 0; d < HD; ++d) m += vv[d] * M[d * HD + f];
    atomicAdd(&aggp[f], m * alpha);
  }
}

__global__ void k_gelu(const float* __restrict__ agg, _Float16* __restrict__ gh, int n) {
  int i = blockIdx.x * 256 + threadIdx.x;
  if (i < n) {
    float x = agg[i];
    float g = 0.5f * x * (1.f + erff(x * 0.70710678118f));
    gh[i] = (_Float16)g;
  }
}

extern "C" void kernel_launch(void* const* d_in, const int* in_sizes, int n_in,
                              void* d_out, int out_size, void* d_ws, size_t ws_size,
                              hipStream_t stream) {
  const float* node_feature = (const float*)d_in[0];
  const int*   node_type    = (const int*)d_in[1];
  const int*   edge_index   = (const int*)d_in[3];
  const int*   edge_type    = (const int*)d_in[4];
  const float* adapt_W      = (const float*)d_in[5];
  const float* adapt_b      = (const float*)d_in[6];
  const float* Wk   = (const float*)d_in[7];
  const float* bk   = (const float*)d_in[8];
  const float* Wq   = (const float*)d_in[9];
  const float* bq   = (const float*)d_in[10];
  const float* Wv   = (const float*)d_in[11];
  const float* bv   = (const float*)d_in[12];
  const float* Wagg = (const float*)d_in[13];
  const float* bagg = (const float*)d_in[14];
  const float* rel_pri = (const float*)d_in[15];
  const float* rel_att = (const float*)d_in[16];
  const float* rel_msg = (const float*)d_in[17];
  const float* skip    = (const float*)d_in[18];

  const int* src = edge_index;
  const int* tgt = edge_index + EE;

  // workspace carve-up
  size_t off = 0;
  auto carve = [&](size_t bytes) -> void* {
    void* p = (char*)d_ws + off;
    off += (bytes + 255) & ~(size_t)255;
    return p;
  };
  _Float16* nfh   = (_Float16*)carve((size_t)NN * D * 2);
  _Float16* xh    = (_Float16*)carve((size_t)NN * D * 2);
  float*    xf    = (float*)   carve((size_t)NN * D * 4);
  float*    kb    = (float*)   carve((size_t)NN * D * 4);
  float*    qb    = (float*)   carve((size_t)NN * D * 4);
  float*    vb    = (float*)   carve((size_t)NN * D * 4);
  float*    attb  = (float*)   carve((size_t)EE * NH * 4);
  unsigned* nmax  = (unsigned*)carve((size_t)NN * NH * 4);
  float*    den   = (float*)   carve((size_t)NN * NH * 4);
  float*    agg   = (float*)   carve((size_t)NN * D * 4);
  _Float16* gh    = (_Float16*)carve((size_t)NN * D * 2);
  _Float16* adWh  = (_Float16*)carve((size_t)NTY * D * D * 2);
  _Float16* Wkh   = (_Float16*)carve((size_t)2 * NTY * D * D * 2);
  _Float16* Wqh   = (_Float16*)carve((size_t)2 * NTY * D * D * 2);
  _Float16* Wvh   = (_Float16*)carve((size_t)2 * NTY * D * D * 2);
  _Float16* Waggh = (_Float16*)carve((size_t)2 * NTY * D * D * 2);

  const int nND = NN * D;                 // 6,400,000
  const int nW1 = NTY * D * D;            // 49,152
  const int nW2 = 2 * NTY * D * D;        // 98,304

  // feature conversion + weight conversion-and-packing into WMMA B-fragment layout
  k_f32_to_f16<<<nND / 256, 256, 0, stream>>>(node_feature, nfh, nND);
  k_pack_w<<<(nW1 + 255) / 256, 256, 0, stream>>>(adapt_W, adWh, NTY);
  k_pack_w<<<(nW2 + 255) / 256, 256, 0, stream>>>(Wk,   Wkh,   2 * NTY);
  k_pack_w<<<(nW2 + 255) / 256, 256, 0, stream>>>(Wq,   Wqh,   2 * NTY);
  k_pack_w<<<(nW2 + 255) / 256, 256, 0, stream>>>(Wv,   Wvh,   2 * NTY);
  k_pack_w<<<(nW2 + 255) / 256, 256, 0, stream>>>(Wagg, Waggh, 2 * NTY);

  // input adaptation: x = tanh(type_linear(node_feature))
  k_gemm_ts<0><<<3125, 256, 0, stream>>>(nfh, adWh, adapt_b, node_type, xf, xh, nullptr, nullptr);

  for (int l = 0; l < 2; ++l) {
    const size_t wo = (size_t)l * NTY * D * D;   // packed layout keeps [mat][...] outermost
    const size_t bo = (size_t)l * NTY * D;

    k_gemm_ts<1><<<3125, 256, 0, stream>>>(xh, Wkh + wo, bk + bo, node_type, kb, nullptr, nullptr, nullptr);
    k_gemm_ts<1><<<3125, 256, 0, stream>>>(xh, Wqh + wo, bq + bo, node_type, qb, nullptr, nullptr, nullptr);
    k_gemm_ts<1><<<3125, 256, 0, stream>>>(xh, Wvh + wo, bv + bo, node_type, vb, nullptr, nullptr, nullptr);

    const int nNH = NN * NH;  // 400,000
    k_fill_u32<<<(nNH + 255) / 256, 256, 0, stream>>>(nmax, KEY_NEG_INF, nNH);
    k_fill_u32<<<(nNH + 255) / 256, 256, 0, stream>>>((unsigned*)den, 0u, nNH);
    k_fill_u32<<<nND / 256, 256, 0, stream>>>((unsigned*)agg, 0u, nND);

    const float* ra = rel_att + (size_t)l * NET * NH * HD * HD;
    const float* rm = rel_msg + (size_t)l * NET * NH * HD * HD;
    const float* rp = rel_pri + (size_t)l * NET * NH;

    k_edge_att<<<(EE * NH) / 256, 256, 0, stream>>>(kb, qb, src, tgt, edge_type, ra, rp, attb, nmax);
    k_edge_ex<<<(EE * NH) / 256, 256, 0, stream>>>(attb, tgt, nmax, den);
    k_edge_scatter<<<(EE * NH) / 256, 256, 0, stream>>>(vb, attb, src, tgt, edge_type, rm, den, agg);

    k_gelu<<<nND / 256, 256, 0, stream>>>(agg, gh, nND);

    float* outp = (l == 1) ? (float*)d_out : xf;
    k_gemm_ts<2><<<3125, 256, 0, stream>>>(gh, Waggh + wo, bagg + bo, node_type, outp, xh, xf, skip + (size_t)l * NTY);
  }
}
